// LaBVaTr_1486058684730
// MI455X (gfx1250) — matrix operand order repaired and checked by reading, not compiled, not run on hardware
//
#include <hip/hip_runtime.h>
#include <math.h>

// ---------------- CDNA5 WMMA plumbing (wave32, 16x16x32 f16 -> f32) ----------------
typedef __attribute__((ext_vector_type(16))) _Float16 v16h;
typedef __attribute__((ext_vector_type(8)))  _Float16 v8h;
typedef __attribute__((ext_vector_type(8)))  float    v8f;

#define NPTS 131072
#define MTOK 4096

// A fragment: 16(M) x 32(K), row-major source with row stride lda (halfs).
// lane l -> m = l&15, half = l>>4; two contiguous 8-half chunks.
__device__ inline v16h load_a16(const _Float16* A, int lda, int k0) {
  int l = threadIdx.x & 31, mn = l & 15, hf = l >> 4;
  union { v16h v; v8h h[2]; } a;
  a.h[0] = *(const v8h*)(A + (size_t)mn * lda + k0 + hf * 8);
  a.h[1] = *(const v8h*)(A + (size_t)mn * lda + k0 + 16 + hf * 8);
  return a.v;
}

// B fragment: 32(K) x 16(N). Source is Bt = B^T row-major [n][k], stride ldb.
__device__ inline v16h load_b16(const _Float16* Bt, int ldb, int n0, int k0) {
  int l = threadIdx.x & 31, mn = l & 15, hf = l >> 4;
  union { v16h v; v8h h[2]; } b;
  b.h[0] = *(const v8h*)(Bt + (size_t)(n0 + mn) * ldb + k0 + hf * 8);
  b.h[1] = *(const v8h*)(Bt + (size_t)(n0 + mn) * ldb + k0 + 16 + hf * 8);
  return b.v;
}

__device__ inline v8f wmma_f16(v16h a, v16h b, v8f c) {
  return __builtin_amdgcn_wmma_f32_16x16x32_f16(false, a, false, b, (short)0, c, false, false);
}

// Hoist all K-step A fragments of a 16-row LDS tile into registers.
template <int KT>
__device__ inline void load_afrags(v16h* af, const _Float16* A, int lda) {
  #pragma unroll
  for (int kt = 0; kt < KT; ++kt) af[kt] = load_a16(A, lda, kt * 32);
}

// Two adjacent n-tiles with shared (register-resident) A fragments.
// Software-pipelined: B fragments for step kt+1 are issued before the two
// wmma of step kt, keeping >=2 loads in flight past each wait.
template <int KT>
__device__ inline void gemm_pair(const v16h* af, const _Float16* Bt, int ldb, int n0,
                                 v8f& c0, v8f& c1) {
  v16h b0 = load_b16(Bt, ldb, n0, 0);
  v16h b1 = load_b16(Bt, ldb, n0 + 16, 0);
  #pragma unroll
  for (int kt = 0; kt < KT; ++kt) {
    v16h nb0 = b0, nb1 = b1;
    if (kt + 1 < KT) {
      nb0 = load_b16(Bt, ldb, n0, (kt + 1) * 32);
      nb1 = load_b16(Bt, ldb, n0 + 16, (kt + 1) * 32);
    }
    c0 = wmma_f16(af[kt], b0, c0);
    c1 = wmma_f16(af[kt], b1, c1);
    b0 = nb0; b1 = nb1;
  }
}

__device__ inline float wave_sum(float v) {
  #pragma unroll
  for (int o = 16; o > 0; o >>= 1) v += __shfl_xor(v, o, 32);
  return v;
}

// ---------------- weight prep: fp32 [K][N] -> f16 transposed [N][Kp] (zero pad) ----
__global__ void k_wtrans(const float* __restrict__ W, _Float16* __restrict__ Wt,
                         int K, int N, int Kp) {
  int i = blockIdx.x * 256 + threadIdx.x;
  if (i >= N * Kp) return;
  int n = i / Kp, k = i % Kp;
  Wt[i] = (k < K) ? (_Float16)W[(size_t)k * N + n] : (_Float16)0.f;
}

// ---------------- pooling tokenizer: 1 wave per token (16 contiguous edges) -------
__global__ void __launch_bounds__(32, 1)
k_pool(const float* __restrict__ x, const float* __restrict__ pos,
       const int* __restrict__ samp, const int* __restrict__ psrc,
       const float* __restrict__ w1, const float* __restrict__ b1,
       const _Float16* __restrict__ w2t, const float* __restrict__ b2,
       const float* __restrict__ g, const float* __restrict__ be,
       float* __restrict__ tok) {
  __shared__ __align__(32) _Float16 h1[16 * 512];
  __shared__ float raw[16 * 128];
  __shared__ float msg[16 * 19];
  int m = blockIdx.x, l = threadIdx.x, n = l & 15, hf = l >> 4;
  int sp = samp[m];
  float px = pos[sp * 3 + 0], py = pos[sp * 3 + 1], pz = pos[sp * 3 + 2];
  for (int t = l; t < 16 * 19; t += 32) {
    int j = t / 19, f = t % 19;
    int s = psrc[m * 16 + j];
    float v;
    if (f < 16) v = x[(size_t)s * 16 + f];
    else        v = pos[s * 3 + (f - 16)] - (f == 16 ? px : (f == 17 ? py : pz));
    msg[j * 19 + f] = v;
  }
  __syncthreads();
  for (int t = l; t < 16 * 512; t += 32) {          // l1: 19 -> 512, relu (VALU)
    int j = t >> 9, o = t & 511;
    float acc = b1[o];
    #pragma unroll
    for (int i = 0; i < 19; ++i) acc += msg[j * 19 + i] * w1[i * 512 + o];
    h1[j * 512 + o] = (_Float16)fmaxf(acc, 0.f);
  }
  __syncthreads();
  {                                                 // l2: 512 -> 128 (WMMA)
    v16h af[16];
    load_afrags<16>(af, h1, 512);
    for (int nt = 0; nt < 8; nt += 2) {
      v8f c0 = {}, c1 = {};
      gemm_pair<16>(af, w2t, 512, nt * 16, c0, c1);
      #pragma unroll
      for (int r = 0; r < 8; ++r) {
        raw[(r + 8 * hf) * 128 + nt * 16 + n] = c0[r] + b2[nt * 16 + n];
        raw[(r + 8 * hf) * 128 + nt * 16 + 16 + n] = c1[r] + b2[nt * 16 + 16 + n];
      }
    }
  }
  __syncthreads();
  for (int j = 0; j < 16; ++j) {                    // LN + relu per edge row
    float s = 0.f;
    for (int c = l; c < 128; c += 32) s += raw[j * 128 + c];
    float mean = wave_sum(s) * (1.f / 128.f);
    float v = 0.f;
    for (int c = l; c < 128; c += 32) { float d = raw[j * 128 + c] - mean; v += d * d; }
    float rstd = rsqrtf(wave_sum(v) * (1.f / 128.f) + 1e-5f);
    for (int c = l; c < 128; c += 32)
      raw[j * 128 + c] = fmaxf((raw[j * 128 + c] - mean) * rstd * g[c] + be[c], 0.f);
  }
  __syncthreads();
  for (int c = l; c < 128; c += 32) {               // segment mean (count == 16)
    float s = 0.f;
    for (int j = 0; j < 16; ++j) s += raw[j * 128 + c];
    tok[(size_t)m * 128 + c] = s * (1.f / 16.f);
  }
}

// ---------------- out16x128 = A16x128 @ Wt + bias (t_in / t_out) ------------------
__global__ void __launch_bounds__(32, 1)
k_dense16(const float* __restrict__ Ain, const _Float16* __restrict__ Wt,
          const float* __restrict__ bias, float* __restrict__ Out) {
  __shared__ __align__(32) _Float16 Ah[16 * 128];
  int t0 = blockIdx.x * 16, l = threadIdx.x, n = l & 15, hf = l >> 4;
  for (int i = l; i < 16 * 128; i += 32) Ah[i] = (_Float16)Ain[(size_t)t0 * 128 + i];
  __syncthreads();
  v16h af[4];
  load_afrags<4>(af, Ah, 128);
  for (int nt = 0; nt < 8; nt += 2) {
    v8f c0 = {}, c1 = {};
    gemm_pair<4>(af, Wt, 128, nt * 16, c0, c1);
    #pragma unroll
    for (int r = 0; r < 8; ++r) {
      Out[(size_t)(t0 + r + 8 * hf) * 128 + nt * 16 + n] = c0[r] + bias[nt * 16 + n];
      Out[(size_t)(t0 + r + 8 * hf) * 128 + nt * 16 + 16 + n] = c1[r] + bias[nt * 16 + 16 + n];
    }
  }
}

// ---------------- LN1 + QKV projections (multi-query: shared K/V head) ------------
__global__ void __launch_bounds__(32, 1)
k_ln_qkv(const float* __restrict__ tin, const float* __restrict__ g1,
         const float* __restrict__ b1v,
         const _Float16* __restrict__ wqt, const float* __restrict__ bq,
         const _Float16* __restrict__ wkt, const float* __restrict__ bkv,
         const _Float16* __restrict__ wvt, const float* __restrict__ bvv,
         _Float16* __restrict__ qh, _Float16* __restrict__ kkh,
         _Float16* __restrict__ vth, int M) {
  __shared__ float row[16 * 128];
  __shared__ __align__(32) _Float16 y[16 * 128];
  int t0 = blockIdx.x * 16, l = threadIdx.x, n = l & 15, hf = l >> 4;
  for (int i = l; i < 16 * 128; i += 32) row[i] = tin[(size_t)t0 * 128 + i];
  __syncthreads();
  for (int j = 0; j < 16; ++j) {
    float s = 0.f;
    for (int c = l; c < 128; c += 32) s += row[j * 128 + c];
    float mean = wave_sum(s) * (1.f / 128.f);
    float v = 0.f;
    for (int c = l; c < 128; c += 32) { float d = row[j * 128 + c] - mean; v += d * d; }
    float rstd = rsqrtf(wave_sum(v) * (1.f / 128.f) + 1e-5f);
    for (int c = l; c < 128; c += 32)
      y[j * 128 + c] = (_Float16)((row[j * 128 + c] - mean) * rstd * g1[c] + b1v[c]);
  }
  __syncthreads();
  v16h af[4];
  load_afrags<4>(af, y, 128);
  for (int nt = 0; nt < 8; nt += 2) {               // Q: [M,128] f16
    v8f c0 = {}, c1 = {};
    gemm_pair<4>(af, wqt, 128, nt * 16, c0, c1);
    #pragma unroll
    for (int r = 0; r < 8; ++r) {
      qh[(size_t)(t0 + r + 8 * hf) * 128 + nt * 16 + n] =
          (_Float16)(c0[r] + bq[nt * 16 + n]);
      qh[(size_t)(t0 + r + 8 * hf) * 128 + nt * 16 + 16 + n] =
          (_Float16)(c1[r] + bq[nt * 16 + 16 + n]);
    }
  }
  {                                                 // K: [M,32] f16 (row-major = B^T for scores)
    v8f c0 = {}, c1 = {};
    gemm_pair<4>(af, wkt, 128, 0, c0, c1);
    #pragma unroll
    for (int r = 0; r < 8; ++r) {
      kkh[(size_t)(t0 + r + 8 * hf) * 32 + n] = (_Float16)(c0[r] + bkv[n]);
      kkh[(size_t)(t0 + r + 8 * hf) * 32 + 16 + n] = (_Float16)(c1[r] + bkv[16 + n]);
    }
  }
  {                                                 // V^T: [32,M] f16 (= B^T for AV)
    v8f c0 = {}, c1 = {};
    gemm_pair<4>(af, wvt, 128, 0, c0, c1);
    #pragma unroll
    for (int r = 0; r < 8; ++r) {
      vth[(size_t)n * M + t0 + r + 8 * hf] = (_Float16)(c0[r] + bvv[n]);
      vth[(size_t)(16 + n) * M + t0 + r + 8 * hf] = (_Float16)(c1[r] + bvv[16 + n]);
    }
  }
}

// ---------------- flash attention: 1 wave per (16-query tile, head) ---------------
__global__ void __launch_bounds__(32, 1)
k_attn(const _Float16* __restrict__ qh, const _Float16* __restrict__ kkh,
       const _Float16* __restrict__ vth, float* __restrict__ attn, int M) {
  __shared__ __align__(32) _Float16 P[16 * 32];
  int qt = blockIdx.x >> 2, h = blockIdx.x & 3;
  int l = threadIdx.x, n = l & 15, hf = l >> 4;
  v16h qa = load_a16(qh + (size_t)qt * 16 * 128 + h * 32, 128, 0);
  float rmax[8], rsum[8];
  v8f O0 = {}, O1 = {};
  #pragma unroll
  for (int r = 0; r < 8; ++r) { rmax[r] = -3.0e38f; rsum[r] = 0.f; }
  const float scale = 0.17677669529663687f;  // 1/sqrt(32)
  for (int kp = 0; kp < M / 32; ++kp) {
    v8f s0 = {}, s1 = {};
    s0 = wmma_f16(qa, load_b16(kkh, 32, kp * 32, 0), s0);
    s1 = wmma_f16(qa, load_b16(kkh, 32, kp * 32 + 16, 0), s1);
    float fac[8];
    #pragma unroll
    for (int r = 0; r < 8; ++r) {                   // online softmax, row = r + 8*hf
      float v0 = s0[r] * scale, v1 = s1[r] * scale;
      float cm = fmaxf(v0, v1);
      #pragma unroll
      for (int o = 1; o < 16; o <<= 1) cm = fmaxf(cm, __shfl_xor(cm, o, 32));
      float nm = fmaxf(rmax[r], cm);
      float p0 = __expf(v0 - nm), p1 = __expf(v1 - nm);
      float cs = p0 + p1;
      #pragma unroll
      for (int o = 1; o < 16; o <<= 1) cs += __shfl_xor(cs, o, 32);
      fac[r] = __expf(rmax[r] - nm);
      rsum[r] = rsum[r] * fac[r] + cs;
      rmax[r] = nm;
      int row = r + 8 * hf;
      P[row * 32 + n] = (_Float16)p0;
      P[row * 32 + 16 + n] = (_Float16)p1;
    }
    #pragma unroll
    for (int r = 0; r < 8; ++r) { O0[r] *= fac[r]; O1[r] *= fac[r]; }
    __syncthreads();
    v16h pa = load_a16(P, 32, 0);
    O0 = wmma_f16(pa, load_b16(vth + kp * 32, M, 0, 0), O0);
    O1 = wmma_f16(pa, load_b16(vth + kp * 32, M, 16, 0), O1);
    __syncthreads();
  }
  #pragma unroll
  for (int r = 0; r < 8; ++r) {
    float inv = 1.f / rsum[r];
    size_t row = (size_t)(qt * 16 + r + 8 * hf);
    attn[row * 128 + h * 32 + n] = O0[r] * inv;
    attn[row * 128 + h * 32 + 16 + n] = O1[r] * inv;
  }
}

// ---------------- o-proj + residual + LN2 + MLP(gelu) + residual ------------------
__global__ void __launch_bounds__(32, 1)
k_oproj_mlp(const float* __restrict__ tin, const float* __restrict__ attn,
            const _Float16* __restrict__ wot, const float* __restrict__ bo,
            const float* __restrict__ g2, const float* __restrict__ b2v,
            const _Float16* __restrict__ m1t, const float* __restrict__ bm1,
            const _Float16* __restrict__ m2t, const float* __restrict__ bm2,
            float* __restrict__ tout) {
  __shared__ __align__(32) _Float16 a16[16 * 128];
  __shared__ float tl[16 * 128];
  __shared__ __align__(32) _Float16 hbuf[16 * 512];
  int t0 = blockIdx.x * 16, l = threadIdx.x, n = l & 15, hf = l >> 4;
  for (int i = l; i < 16 * 128; i += 32) a16[i] = (_Float16)attn[(size_t)t0 * 128 + i];
  for (int i = l; i < 16 * 128; i += 32) tl[i] = tin[(size_t)t0 * 128 + i];
  __syncthreads();
  {                                                 // t += attn @ Wo + bo
    v16h af[4];
    load_afrags<4>(af, a16, 128);
    for (int nt = 0; nt < 8; nt += 2) {
      v8f c0 = {}, c1 = {};
      gemm_pair<4>(af, wot, 128, nt * 16, c0, c1);
      #pragma unroll
      for (int r = 0; r < 8; ++r) {
        tl[(r + 8 * hf) * 128 + nt * 16 + n] += c0[r] + bo[nt * 16 + n];
        tl[(r + 8 * hf) * 128 + nt * 16 + 16 + n] += c1[r] + bo[nt * 16 + 16 + n];
      }
    }
  }
  __syncthreads();
  for (int j = 0; j < 16; ++j) {                    // y = LN2(t) -> f16
    float s = 0.f;
    for (int c = l; c < 128; c += 32) s += tl[j * 128 + c];
    float mean = wave_sum(s) * (1.f / 128.f);
    float v = 0.f;
    for (int c = l; c < 128; c += 32) { float d = tl[j * 128 + c] - mean; v += d * d; }
    float rstd = rsqrtf(wave_sum(v) * (1.f / 128.f) + 1e-5f);
    for (int c = l; c < 128; c += 32)
      a16[j * 128 + c] = (_Float16)((tl[j * 128 + c] - mean) * rstd * g2[c] + b2v[c]);
  }
  __syncthreads();
  {                                                 // h = gelu(y @ m1 + b)
    v16h af[4];
    load_afrags<4>(af, a16, 128);
    for (int nt = 0; nt < 32; nt += 2) {
      v8f c0 = {}, c1 = {};
      gemm_pair<4>(af, m1t, 128, nt * 16, c0, c1);
      #pragma unroll
      for (int r = 0; r < 8; ++r) {
        float v0 = c0[r] + bm1[nt * 16 + n];
        float v1 = c1[r] + bm1[nt * 16 + 16 + n];
        float g0 = 0.5f * v0 * (1.f + tanhf(0.7978845608f * (v0 + 0.044715f * v0 * v0 * v0)));
        float g1v = 0.5f * v1 * (1.f + tanhf(0.7978845608f * (v1 + 0.044715f * v1 * v1 * v1)));
        hbuf[(r + 8 * hf) * 512 + nt * 16 + n] = (_Float16)g0;
        hbuf[(r + 8 * hf) * 512 + nt * 16 + 16 + n] = (_Float16)g1v;
      }
    }
  }
  __syncthreads();
  {                                                 // t = t + h @ m2 + b
    v16h af[16];
    load_afrags<16>(af, hbuf, 512);
    for (int nt = 0; nt < 8; nt += 2) {
      v8f c0 = {}, c1 = {};
      gemm_pair<16>(af, m2t, 512, nt * 16, c0, c1);
      #pragma unroll
      for (int r = 0; r < 8; ++r) {
        tout[(size_t)(t0 + r + 8 * hf) * 128 + nt * 16 + n] =
            tl[(r + 8 * hf) * 128 + nt * 16 + n] + c0[r] + bm2[nt * 16 + n];
        tout[(size_t)(t0 + r + 8 * hf) * 128 + nt * 16 + 16 + n] =
            tl[(r + 8 * hf) * 128 + nt * 16 + 16 + n] + c1[r] + bm2[nt * 16 + 16 + n];
      }
    }
  }
}

// ---------------- fused lift: IDW interp + MLP(144->512->512->4) ------------------
__global__ void __launch_bounds__(32, 1)
k_lift(const float* __restrict__ x, const float* __restrict__ pos,
       const int* __restrict__ samp, const int* __restrict__ isrc,
       const float* __restrict__ tfin,
       const _Float16* __restrict__ l1t, const float* __restrict__ b1,
       const _Float16* __restrict__ l2t, const float* __restrict__ b2,
       const float* __restrict__ g, const float* __restrict__ be,
       const float* __restrict__ w3, const float* __restrict__ b3,
       float* __restrict__ out) {
  __shared__ __align__(32) _Float16 A[16 * 160];
  __shared__ __align__(32) _Float16 h1[16 * 512];
  __shared__ float raw[16 * 512];
  int p0 = blockIdx.x * 16, l = threadIdx.x, n = l & 15, hf = l >> 4;
  for (int j = 0; j < 16; ++j) {                    // z = [IDW(tfin), x], pad to 160
    int p = p0 + j;
    float qx = pos[p * 3], qy = pos[p * 3 + 1], qz = pos[p * 3 + 2];
    float wj[4]; int sj[4];
    #pragma unroll
    for (int k = 0; k < 4; ++k) {
      int s = isrc[p * 4 + k]; sj[k] = s;
      int sp = samp[s];
      float dx = pos[sp * 3] - qx, dy = pos[sp * 3 + 1] - qy, dz = pos[sp * 3 + 2] - qz;
      wj[k] = 1.f / fmaxf(dx * dx + dy * dy + dz * dz, 1e-16f);
    }
    float inv = 1.f / (wj[0] + wj[1] + wj[2] + wj[3]);
    for (int c = l; c < 128; c += 32) {
      float acc = 0.f;
      #pragma unroll
      for (int k = 0; k < 4; ++k) acc += wj[k] * tfin[(size_t)sj[k] * 128 + c];
      A[j * 160 + c] = (_Float16)(acc * inv);
    }
    int cc = 128 + l;  // l in [0,32): fills cols 128..159
    A[j * 160 + cc] = (cc < 144) ? (_Float16)x[(size_t)p * 16 + (cc - 128)] : (_Float16)0.f;
  }
  __syncthreads();
  {                                                 // l1 (K=160 padded), relu
    v16h af[5];
    load_afrags<5>(af, A, 160);
    for (int nt = 0; nt < 32; nt += 2) {
      v8f c0 = {}, c1 = {};
      gemm_pair<5>(af, l1t, 160, nt * 16, c0, c1);
      #pragma unroll
      for (int r = 0; r < 8; ++r) {
        h1[(r + 8 * hf) * 512 + nt * 16 + n] = (_Float16)fmaxf(c0[r] + b1[nt * 16 + n], 0.f);
        h1[(r + 8 * hf) * 512 + nt * 16 + 16 + n] =
            (_Float16)fmaxf(c1[r] + b1[nt * 16 + 16 + n], 0.f);
      }
    }
  }
  __syncthreads();
  {                                                 // l2 raw
    v16h af[16];
    load_afrags<16>(af, h1, 512);
    for (int nt = 0; nt < 32; nt += 2) {
      v8f c0 = {}, c1 = {};
      gemm_pair<16>(af, l2t, 512, nt * 16, c0, c1);
      #pragma unroll
      for (int r = 0; r < 8; ++r) {
        raw[(r + 8 * hf) * 512 + nt * 16 + n] = c0[r] + b2[nt * 16 + n];
        raw[(r + 8 * hf) * 512 + nt * 16 + 16 + n] = c1[r] + b2[nt * 16 + 16 + n];
      }
    }
  }
  __syncthreads();
  for (int j = 0; j < 16; ++j) {                    // LN + relu -> h2 (reuse h1)
    float s = 0.f;
    for (int c = l; c < 512; c += 32) s += raw[j * 512 + c];
    float mean = wave_sum(s) * (1.f / 512.f);
    float v = 0.f;
    for (int c = l; c < 512; c += 32) { float d = raw[j * 512 + c] - mean; v += d * d; }
    float rstd = rsqrtf(wave_sum(v) * (1.f / 512.f) + 1e-5f);
    for (int c = l; c < 512; c += 32)
      h1[j * 512 + c] = (_Float16)fmaxf((raw[j * 512 + c] - mean) * rstd * g[c] + be[c], 0.f);
  }
  __syncthreads();
  {                                                 // l3: 512 -> 4 (VALU)
    int r = l & 15;
    for (int ci = 0; ci < 2; ++ci) {
      int c = (l >> 4) + ci * 2;
      float acc = b3[c];
      for (int k = 0; k < 512; ++k) acc += (float)h1[r * 512 + k] * w3[k * 4 + c];
      out[(size_t)(p0 + r) * 4 + c] = acc;
    }
  }
}

// ---------------- host side ----------------
static void launch_tr(const float* W, _Float16* Wt, int K, int N, int Kp, hipStream_t s) {
  int total = N * Kp;
  k_wtrans<<<(total + 255) / 256, 256, 0, s>>>(W, Wt, K, N, Kp);
}

extern "C" void kernel_launch(void* const* d_in, const int* in_sizes, int n_in,
                              void* d_out, int out_size, void* d_ws, size_t ws_size,
                              hipStream_t stream) {
  (void)in_sizes; (void)n_in; (void)out_size; (void)ws_size;
  // Input order: setup_inputs() insertion order; params pytree flattened
  // recursively in insertion order (dense dicts: w then b).
  const float* x    = (const float*)d_in[0];
  const float* pos  = (const float*)d_in[1];
  const int*   samp = (const int*)d_in[2];
  const int*   psrc = (const int*)d_in[3];
  /* d_in[4] pool_tgt: contiguous repeat, implicit */
  const int*   isrc = (const int*)d_in[5];
  /* d_in[6] interp_tgt: contiguous repeat, implicit */
  const float* pl1w = (const float*)d_in[7];
  const float* pl1b = (const float*)d_in[8];
  const float* pl2w = (const float*)d_in[9];
  const float* pl2b = (const float*)d_in[10];
  const float* plng = (const float*)d_in[11];
  const float* plnb = (const float*)d_in[12];
  const float* ll1w = (const float*)d_in[13];
  const float* ll1b = (const float*)d_in[14];
  const float* ll2w = (const float*)d_in[15];
  const float* ll2b = (const float*)d_in[16];
  const float* llng = (const float*)d_in[17];
  const float* llnb = (const float*)d_in[18];
  const float* ll3w = (const float*)d_in[19];
  const float* ll3b = (const float*)d_in[20];
  const float* tinw = (const float*)d_in[21];
  const float* tinb = (const float*)d_in[22];
  const float* toutw = (const float*)d_in[23];
  const float* toutb = (const float*)d_in[24];

  // workspace carve
  char* wsp = (char*)d_ws; size_t off = 0;
  #define CARVE(T, nbytes) (T*)(wsp + off); off = (off + (size_t)(nbytes) + 255) & ~(size_t)255
  float* tok  = CARVE(float, MTOK * 128 * 4);
  float* tbuf = CARVE(float, MTOK * 128 * 4);
  float* tfin = CARVE(float, MTOK * 128 * 4);
  float* attn = CARVE(float, MTOK * 128 * 4);
  _Float16* qh  = CARVE(_Float16, MTOK * 128 * 2);
  _Float16* kkh = CARVE(_Float16, MTOK * 32 * 2);
  _Float16* vth = CARVE(_Float16, 32 * MTOK * 2);
  _Float16* w2t  = CARVE(_Float16, 128 * 512 * 2);
  _Float16* tint = CARVE(_Float16, 128 * 128 * 2);
  _Float16* toutt = CARVE(_Float16, 128 * 128 * 2);
  _Float16* l1t = CARVE(_Float16, 512 * 160 * 2);
  _Float16* l2t = CARVE(_Float16, 512 * 512 * 2);
  _Float16 *wqt[4], *wkt[4], *wvt[4], *wot[4], *m1t[4], *m2t[4];
  for (int b = 0; b < 4; ++b) {
    wqt[b] = CARVE(_Float16, 128 * 128 * 2);
    wkt[b] = CARVE(_Float16, 32 * 128 * 2);
    wvt[b] = CARVE(_Float16, 32 * 128 * 2);
    wot[b] = CARVE(_Float16, 128 * 128 * 2);
    m1t[b] = CARVE(_Float16, 512 * 128 * 2);
    m2t[b] = CARVE(_Float16, 128 * 512 * 2);
  }
  #undef CARVE

  // weight prep (f16, transposed, zero-padded K)
  launch_tr(pl2w, w2t, 512, 128, 512, stream);
  launch_tr(tinw, tint, 128, 128, 128, stream);
  launch_tr(toutw, toutt, 128, 128, 128, stream);
  launch_tr(ll1w, l1t, 144, 512, 160, stream);
  launch_tr(ll2w, l2t, 512, 512, 512, stream);
  for (int b = 0; b < 4; ++b) {
    const float* const* bp = (const float* const*)(d_in + 25 + 16 * b);
    launch_tr(bp[2],  wqt[b], 128, 128, 128, stream);
    launch_tr(bp[4],  wkt[b], 128, 32, 128, stream);
    launch_tr(bp[6],  wvt[b], 128, 32, 128, stream);
    launch_tr(bp[8],  wot[b], 128, 128, 128, stream);
    launch_tr(bp[12], m1t[b], 128, 512, 128, stream);
    launch_tr(bp[14], m2t[b], 512, 128, 512, stream);
  }

  // pipeline
  k_pool<<<MTOK, 32, 0, stream>>>(x, pos, samp, psrc, pl1w, pl1b, w2t, pl2b, plng, plnb, tok);
  k_dense16<<<MTOK / 16, 32, 0, stream>>>(tok, tint, tinb, tbuf);
  for (int b = 0; b < 4; ++b) {
    const float* const* bp = (const float* const*)(d_in + 25 + 16 * b);
    k_ln_qkv<<<MTOK / 16, 32, 0, stream>>>(tbuf, bp[0], bp[1], wqt[b], bp[3], wkt[b], bp[5],
                                           wvt[b], bp[7], qh, kkh, vth, MTOK);
    k_attn<<<(MTOK / 16) * 4, 32, 0, stream>>>(qh, kkh, vth, attn, MTOK);
    k_oproj_mlp<<<MTOK / 16, 32, 0, stream>>>(tbuf, attn, wot[b], bp[9], bp[10], bp[11],
                                              m1t[b], bp[13], m2t[b], bp[15], tbuf);
  }
  k_dense16<<<MTOK / 16, 32, 0, stream>>>(tbuf, toutt, toutb, tfin);
  k_lift<<<NPTS / 16, 32, 0, stream>>>(x, pos, samp, isrc, tfin, l1t, ll1b, l2t, ll2b,
                                       llng, llnb, ll3b ? ll3w : ll3w, ll3b, (float*)d_out);
}
